// GraphIsomorphismNetwork_81037442941188
// MI455X (gfx1250) — matrix operand order repaired and checked by reading, not compiled, not run on hardware
//
#include <hip/hip_runtime.h>

typedef __attribute__((ext_vector_type(16))) __bf16 v16bf;
typedef __attribute__((ext_vector_type(8)))  float  v8f;
typedef unsigned int u32x4 __attribute__((ext_vector_type(4)));
typedef int          i32x4 __attribute__((ext_vector_type(4)));
typedef int          i32x8 __attribute__((ext_vector_type(8)));

#define NNODES 50000
#define NEDGES 800000
#define BM 128
#define BN 32
#define BK 32

// ---------------------------------------------------------------- utilities
__global__ void zero_f32(float* __restrict__ p, int n) {
  int i = blockIdx.x * blockDim.x + threadIdx.x;
  if (i < n) p[i] = 0.0f;
}

// convert W [K][Nd] f32 -> Wt [Nd][K] bf16 (transposed so TDM tiles land in
// LDS with K contiguous, matching the WMMA B-fragment pair layout)
__global__ void f32_to_bf16_T(const float* __restrict__ w, __bf16* __restrict__ o,
                              int K, int Nd) {
  int i = blockIdx.x * blockDim.x + threadIdx.x;
  if (i < K * Nd) {
    int k = i / Nd, n = i % Nd;
    o[(long long)n * K + k] = (__bf16)w[i];
  }
}

// ------------------------------------------------- edge scatter-add (segment_sum)
// agg[dst[e]*K + c] += x[src[e]*K + c]; one thread per (edge, channel-pair).
// All destination arrays (<= 51 MB) fit in the 192 MB L2, so the f32 atomics
// resolve at L2 bandwidth, not HBM.
__global__ void gin_aggregate(const float* __restrict__ x,
                              const long long* __restrict__ src,
                              const long long* __restrict__ dst,
                              float* __restrict__ agg, int K) {
  long long t = (long long)blockIdx.x * blockDim.x + threadIdx.x;
  int kh = K >> 1;
  long long total = (long long)NEDGES * kh;
  if (t >= total) return;
  int e = (int)(t / kh);
  int c = (int)(t % kh) * 2;
  __builtin_prefetch((const void*)(src + e + 256), 0, 1);  // global_prefetch_b8
  __builtin_prefetch((const void*)(dst + e + 256), 0, 1);
  int s = (int)src[e];
  int d = (int)dst[e];
  const float2 v = *(const float2*)(x + (long long)s * K + c);
  atomicAdd(agg + (long long)d * K + c,     v.x);   // global_atomic_add_f32
  atomicAdd(agg + (long long)d * K + c + 1, v.y);
}

// ------------------------------------------------- fused GEMM+bias+BN+ReLU+skip
// out = relu( ((x+agg) @ W) * scale + shift ) + skip
// Block = 256 threads = 8 wave32; block tile 128x32. Each wave owns one
// 16-row slice and BOTH 16-col halves (2 accumulators, shared A fragment ->
// 2 WMMAs per K-step). B tile arrives via the Tensor Data Mover (TDM).
__global__ __launch_bounds__(256)
void gin_wmma_gemm(const float* __restrict__ x,     // [M,K]
                   const float* __restrict__ agg,   // [M,K]
                   const __bf16* __restrict__ wBt,  // [Nd,K] bf16 (transposed)
                   const float* __restrict__ bias,
                   const float* __restrict__ gamma,
                   const float* __restrict__ beta,
                   const float* __restrict__ mean,
                   const float* __restrict__ var,
                   const float* __restrict__ skip,  // [M,Nd]
                   float* __restrict__ out,         // [M,Nd]
                   int M, int K, int Nd) {
  __shared__ unsigned short As[BM][BK];   // (x+agg) tile, bf16 bits, [row][k]  (8 KB)
  __shared__ unsigned short Bs[BN][BK];   // Wt tile, [n][k] (K contiguous)    (2 KB)

  const int m0   = blockIdx.x * BM;
  const int n0   = blockIdx.y * BN;
  const int tid  = threadIdx.x;
  const int wave = tid >> 5;
  const int lane = tid & 31;

  v8f acc0 = {}, acc1 = {};

  // loop-invariant TDM descriptor pieces
  const unsigned ldsB = (unsigned)(unsigned long long)&Bs[0][0];  // LDS byte offset (low 32 bits of generic addr)
  const unsigned long long gBase =
      (unsigned long long)(const void*)wBt + 2ull * ((unsigned long long)n0 * (unsigned)K);

  const int ksteps = (K + BK - 1) / BK;
  for (int ks = 0; ks < ksteps; ++ks) {
    const int k0 = ks * BK;

    // ---- B tile via Tensor Data Mover: 2D tile (dim0 = k, BK wide; dim1 = n, BN rows)
    if (wave == 0) {
      const unsigned long long ga = gBase + 2ull * (unsigned)k0;  // tile start
      u32x4 g0;
      g0.x = 1u;                                     // count=1 (valid user D#)
      g0.y = ldsB;                                   // lds_addr
      g0.z = (unsigned)ga;                           // global_addr[31:0]
      g0.w = (unsigned)(ga >> 32) | (2u << 30);      // global_addr[56:32] | type=2
      i32x8 g1;
      g1[0] = 0x00010000;                            // wg_mask=0 | data_size=1 (2B)
      g1[1] = (int)((unsigned)(K & 0xFFFF) << 16);   // tensor_dim0[15:0] (=K)
      g1[2] = (int)((unsigned)(Nd & 0xFFFF) << 16);  // tensor_dim0[31:16]=0 | tensor_dim1[15:0] (=Nd)
      g1[3] = (int)(BK << 16);                       // tensor_dim1[31:16]=0 | tile_dim0=BK
      g1[4] = BN;                                    // tile_dim1=BN | tile_dim2=0
      g1[5] = K;                                     // tensor_dim0_stride[31:0] = K
      g1[6] = 0;                                     // stride hi | tensor_dim1_stride lo
      g1[7] = 0;
      i32x4 gz4 = {0, 0, 0, 0};
#if defined(__clang_major__) && (__clang_major__ >= 23)
      i32x8 gz8 = {0, 0, 0, 0, 0, 0, 0, 0};
      __builtin_amdgcn_tensor_load_to_lds(g0, g1, gz4, gz4, gz8, 0);
#else
      __builtin_amdgcn_tensor_load_to_lds(g0, g1, gz4, gz4, 0);
#endif
    }

    // ---- stage A: 128x32 fp32 -> bf16 in LDS; float2-vectorized (8B aligned for all K)
#pragma unroll
    for (int it = 0; it < (BM * BK) / (256 * 4); ++it) {   // 4 iterations, 4 elems each
      int i = tid + it * 256;
      int row = i >> 3;            // 8 groups of 4 per 32-wide row
      int c = (i & 7) * 4;
      int gm = m0 + row, gk = k0 + c;
      union { __bf16 h[4]; unsigned long long u; } pk;
      if (gm < M && (gk + 3) < K) {
        long long idx = (long long)gm * K + gk;
        const float2 xa0 = *(const float2*)(x + idx);
        const float2 xa1 = *(const float2*)(x + idx + 2);
        const float2 ag0 = *(const float2*)(agg + idx);
        const float2 ag1 = *(const float2*)(agg + idx + 2);
        pk.h[0] = (__bf16)(xa0.x + ag0.x);
        pk.h[1] = (__bf16)(xa0.y + ag0.y);
        pk.h[2] = (__bf16)(xa1.x + ag1.x);
        pk.h[3] = (__bf16)(xa1.y + ag1.y);
      } else {
#pragma unroll
        for (int j = 0; j < 4; ++j) {
          float vv = 0.0f;
          int gkj = gk + j;
          if (gm < M && gkj < K) {
            long long idx = (long long)gm * K + gkj;
            vv = x[idx] + agg[idx];
          }
          pk.h[j] = (__bf16)vv;
        }
      }
      *(unsigned long long*)&As[row][c] = pk.u;            // 8B aligned (c%4==0)
    }

    if (wave == 0) __builtin_amdgcn_s_wait_tensorcnt(0);   // TDM tile landed
    __syncthreads();                                       // publish As + Bs

    // ---- fragments (wave32; contiguous k order k = 16*(lane/16) + element, same in A and B)
    // A: lane L -> row M=L%16; K-pair base 0 (lanes 0-15) / 16 (lanes 16-31);
    //    vgpr v -> K=base+2v
    union { v16bf v; unsigned int u[8]; } afrag, bfrag0, bfrag1;
    const int arow = wave * 16 + (lane & 15);
    const int akb  = (lane < 16) ? 0 : 16;
#pragma unroll
    for (int r = 0; r < 8; ++r) {
      int k = akb + 2 * r;
      afrag.u[r] = *(const unsigned int*)&As[arow][k];
    }
    // B: lane L -> col N=L%16; lanes 0-15 hold K=0..15, lanes 16-31 hold K=16..31
    const int bkb = (lane < 16) ? 0 : 16;
#pragma unroll
    for (int r = 0; r < 8; ++r)
      bfrag0.u[r] = *(const unsigned int*)&Bs[(lane & 15)][bkb + 2 * r];
#pragma unroll
    for (int r = 0; r < 8; ++r)
      bfrag1.u[r] = *(const unsigned int*)&Bs[16 + (lane & 15)][bkb + 2 * r];

    // EXEC is all-ones here; shared A fragment feeds both column halves
    acc0 = __builtin_amdgcn_wmma_f32_16x16x32_bf16(
        false, afrag.v, false, bfrag0.v, (short)0, acc0, false, false);
    acc1 = __builtin_amdgcn_wmma_f32_16x16x32_bf16(
        false, afrag.v, false, bfrag1.v, (short)0, acc1, false, false);
    __syncthreads();                                       // before next stage/TDM
  }

  // ---- epilogue: folded BN per column; C layout: N=lane%16, vgpr r -> row r / r+8
  const int col0 = n0 + (lane & 15);
  const int col1 = col0 + 16;
  const float sc0 = gamma[col0] * rsqrtf(var[col0] + 1e-5f);
  const float sh0 = (bias[col0] - mean[col0]) * sc0 + beta[col0];
  const float sc1 = gamma[col1] * rsqrtf(var[col1] + 1e-5f);
  const float sh1 = (bias[col1] - mean[col1]) * sc1 + beta[col1];
  const int rowbase = m0 + wave * 16 + ((lane < 16) ? 0 : 8);
#pragma unroll
  for (int r = 0; r < 8; ++r) {
    int row = rowbase + r;
    if (row < M) {
      long long ob = (long long)row * Nd;
      float h0 = fmaxf(acc0[r] * sc0 + sh0, 0.0f);
      float h1 = fmaxf(acc1[r] * sc1 + sh1, 0.0f);
      out[ob + col0] = h0 + skip[ob + col0];
      out[ob + col1] = h1 + skip[ob + col1];
    }
  }
}

// ---------------------------------------------------------------- launch
extern "C" void kernel_launch(void* const* d_in, const int* in_sizes, int n_in,
                              void* d_out, int out_size, void* d_ws, size_t ws_size,
                              hipStream_t stream) {
  (void)in_sizes; (void)n_in; (void)out_size; (void)ws_size;

  const float*     v   = (const float*)d_in[0];
  const long long* ei  = (const long long*)d_in[1];   // int64 [2,E]
  const float*     v1  = (const float*)d_in[2];
  const float*     v2  = (const float*)d_in[3];
  const float*     v3  = (const float*)d_in[4];
  const float *w1 = (const float*)d_in[5],  *b1 = (const float*)d_in[6],
              *g1 = (const float*)d_in[7],  *be1= (const float*)d_in[8],
              *m1 = (const float*)d_in[9],  *s1 = (const float*)d_in[10];
  const float *w2 = (const float*)d_in[11], *b2 = (const float*)d_in[12],
              *g2 = (const float*)d_in[13], *be2= (const float*)d_in[14],
              *m2 = (const float*)d_in[15], *s2 = (const float*)d_in[16];
  const float *w3 = (const float*)d_in[17], *b3 = (const float*)d_in[18],
              *g3 = (const float*)d_in[19], *be3= (const float*)d_in[20],
              *m3 = (const float*)d_in[21], *s3 = (const float*)d_in[22];
  const long long* src = ei;
  const long long* dst = ei + NEDGES;
  float* out = (float*)d_out;

  // workspace carve-up
  char* ws = (char*)d_ws;
  size_t off = 0;
  auto carve = [&](size_t bytes) -> void* {
    void* p = ws + off;
    off = (off + bytes + 255) & ~(size_t)255;
    return p;
  };
  __bf16* wb1 = (__bf16*)carve((size_t)86  * 128 * 2);   // [128][86]
  __bf16* wb2 = (__bf16*)carve((size_t)128 * 256 * 2);   // [256][128]
  __bf16* wb3 = (__bf16*)carve((size_t)256 * 512 * 2);   // [512][256]
  float*  agg = (float*) carve((size_t)NNODES * 256 * 4);
  float*  xv1 = (float*) carve((size_t)NNODES * 128 * 4);
  float*  xv2 = (float*) carve((size_t)NNODES * 256 * 4);

  // weights -> bf16, transposed to [Nd][K]
  {
    int n1 = 86 * 128, n2 = 128 * 256, n3 = 256 * 512;
    f32_to_bf16_T<<<(n1 + 255) / 256, 256, 0, stream>>>(w1, wb1, 86,  128);
    f32_to_bf16_T<<<(n2 + 255) / 256, 256, 0, stream>>>(w2, wb2, 128, 256);
    f32_to_bf16_T<<<(n3 + 255) / 256, 256, 0, stream>>>(w3, wb3, 256, 512);
  }

  const int mblocks = (NNODES + BM - 1) / BM;  // 391

  // ---- layer 1: K=86, Nd=128; out = xv1 = relu(BN((v+agg)W1+b1)) + v1
  {
    int K = 86, Nd = 128;
    int nz = NNODES * K;
    zero_f32<<<(nz + 255) / 256, 256, 0, stream>>>(agg, nz);
    long long nt = (long long)NEDGES * (K / 2);
    gin_aggregate<<<(unsigned)((nt + 255) / 256), 256, 0, stream>>>(v, src, dst, agg, K);
    gin_wmma_gemm<<<dim3(mblocks, Nd / BN), 256, 0, stream>>>(
        v, agg, wb1, b1, g1, be1, m1, s1, v1, xv1, NNODES, K, Nd);
  }
  // ---- layer 2: K=128, Nd=256; out = xv2
  {
    int K = 128, Nd = 256;
    int nz = NNODES * K;
    zero_f32<<<(nz + 255) / 256, 256, 0, stream>>>(agg, nz);
    long long nt = (long long)NEDGES * (K / 2);
    gin_aggregate<<<(unsigned)((nt + 255) / 256), 256, 0, stream>>>(xv1, src, dst, agg, K);
    gin_wmma_gemm<<<dim3(mblocks, Nd / BN), 256, 0, stream>>>(
        xv1, agg, wb2, b2, g2, be2, m2, s2, v2, xv2, NNODES, K, Nd);
  }
  // ---- layer 3: K=256, Nd=512; out = d_out = x3 + v3
  {
    int K = 256, Nd = 512;
    int nz = NNODES * K;
    zero_f32<<<(nz + 255) / 256, 256, 0, stream>>>(agg, nz);
    long long nt = (long long)NEDGES * (K / 2);
    gin_aggregate<<<(unsigned)((nt + 255) / 256), 256, 0, stream>>>(xv2, src, dst, agg, K);
    gin_wmma_gemm<<<dim3(mblocks, Nd / BN), 256, 0, stream>>>(
        xv2, agg, wb3, b3, g3, be3, m3, s3, v3, out, NNODES, K, Nd);
  }
}
